// VanillaGAT_69879117906410
// MI455X (gfx1250) — matrix operand order repaired and checked by reading, not compiled, not run on hardware
//
#include <hip/hip_runtime.h>

typedef __attribute__((ext_vector_type(16))) _Float16 v16h;
typedef __attribute__((ext_vector_type(8)))  _Float16 v8h;
typedef __attribute__((ext_vector_type(8)))  float    v8f;
typedef __attribute__((ext_vector_type(4))) unsigned  u32x4;
typedef __attribute__((ext_vector_type(8))) int       i32x8;
typedef __attribute__((ext_vector_type(4))) int       i32x4;

#define NN    50000
#define EE    640000
#define ETOT  (EE + NN)
#define IND   128
#define HIDC  64
#define HEADS 8
#define D1    (HEADS * HIDC)   /* 512 */
#define OUTD  2
#define NG    64
#define NEG_SLOPE 0.2f
#define NEGINF_KEY 0x007FFFFFu /* monotonic key of -inf */

#define HAVE_TDM __has_builtin(__builtin_amdgcn_tensor_load_to_lds)

// ---------- helpers ----------
__device__ __forceinline__ unsigned f2key(float f) {
  unsigned b = __float_as_uint(f);
  return b ^ (unsigned)(((int)b >> 31) | (int)0x80000000);
}
__device__ __forceinline__ float key2f(unsigned k) {
  unsigned b = k ^ ((k & 0x80000000u) ? 0x80000000u : 0xFFFFFFFFu);
  return __uint_as_float(b);
}
__device__ __forceinline__ float leaky(float v) { return v > 0.f ? v : NEG_SLOPE * v; }
__device__ __forceinline__ float elu(float v) { return v > 0.f ? v : (__expf(v) - 1.f); }

// ---------- small utility kernels ----------
__global__ void fill_u32_kernel(unsigned* p, unsigned v, int n) {
  int t = blockIdx.x * blockDim.x + threadIdx.x;
  if (t < n) p[t] = v;
}

__global__ void cast_f16_kernel(const float* __restrict__ s, _Float16* __restrict__ d, int n) {
  int t = blockIdx.x * blockDim.x + threadIdx.x;
  if (t < n) d[t] = (_Float16)s[t];
}

// src is [R x C] row-major f32; dst is [C x R] row-major f16 (i.e. src transposed)
__global__ void transpose_f16_kernel(const float* __restrict__ s, _Float16* __restrict__ d,
                                     int R, int C) {
  int t = blockIdx.x * blockDim.x + threadIdx.x;
  if (t >= R * C) return;
  int r = t / C, c = t % C;
  d[(size_t)c * R + r] = (_Float16)s[t];
}

// ---------- WMMA GEMM: C[M x Ncols] = A[M x K] (f16, row-major) * Bt[Ncols x K] (f16 = B^T) ----
// Block = 256 threads = 8 waves; each wave owns one 16-row strip; the block shares a
// 64-column B panel staged in LDS (padded +8 halves/col for bank-conflict-free b128 reads).
// The panel is loaded by the Tensor Data Mover (TENSOR_LOAD_TO_LDS + s_wait_tensorcnt)
// with pad_enable producing the banked layout directly; cooperative-copy fallback otherwise.
__global__ void wmma_gemm_kernel(const _Float16* __restrict__ A,
                                 const _Float16* __restrict__ Bt,
                                 float* __restrict__ C,
                                 int M, int K, int Ncols) {
  extern __shared__ char smem_raw[];
  _Float16* sB = (_Float16*)smem_raw;          // [64][K + 8] column-major panel
  const int KP = K + 8;

  const int lane = threadIdx.x & 31;
  const int wave = threadIdx.x >> 5;
  const int rowTile = blockIdx.x * (blockDim.x >> 5) + wave;
  const int row0 = rowTile * 16;
  const int col0 = blockIdx.y * 64;
  const int hf = lane >> 4;                    // half-wave selector
  const int lr = lane & 15;

  // ---- stage the 64 x K B panel into LDS ----
#if HAVE_TDM
  if (wave == 0) {
    unsigned lds_base = (unsigned)(size_t)(void*)sB;         // LDS aperture: offset = addr[31:0]
    unsigned long long ga = (unsigned long long)(size_t)(Bt + (size_t)col0 * K);
    // pad_interval: trigger after K halves (= K/2 dwords): 64 dwords -> code 5, 256 -> code 7
    unsigned padi = (K == 128) ? 5u : 7u;
    u32x4 g0;
    g0[0] = 1u;                                              // count=1 (valid), user mode
    g0[1] = lds_base;                                        // lds_addr [63:32]
    g0[2] = (unsigned)(ga & 0xFFFFFFFFu);                    // global_addr[31:0]
    g0[3] = (unsigned)((ga >> 32) & 0x01FFFFFFu) | (2u << 30); // global_addr[56:32] | type=2
    i32x8 g1;
    g1[0] = (int)((1u << 16)            // data_size = 1 (2 bytes)
                | (1u << 20)            // pad_enable
                | (padi << 22)          // pad_interval
                | (3u << 25));          // pad_amount = 4 dwords (8 halves)
    g1[1] = (int)(((unsigned)K & 0xFFFFu) << 16);            // tensor_dim0[15:0]
    g1[2] = (int)((((unsigned)K >> 16) & 0xFFFFu) | (((unsigned)Ncols & 0xFFFFu) << 16));
    g1[3] = (int)(((unsigned)Ncols >> 16) | (((unsigned)K & 0xFFFFu) << 16)); // tile_dim0 = K
    g1[4] = 64;                                              // tile_dim1 = 64, tile_dim2 = 0
    g1[5] = K;                                               // tensor_dim0_stride[31:0]
    g1[6] = 0;                                               // stride hi / dim1_stride lo
    g1[7] = 0;
    i32x4 gz;  gz[0] = 0; gz[1] = 0; gz[2] = 0; gz[3] = 0;   // 2D tensor: groups 2/3 unused
#if __clang_major__ >= 23
    i32x8 z8;  z8[0]=0; z8[1]=0; z8[2]=0; z8[3]=0; z8[4]=0; z8[5]=0; z8[6]=0; z8[7]=0;
    __builtin_amdgcn_tensor_load_to_lds(g0, g1, gz, gz, z8, 0);
#else
    __builtin_amdgcn_tensor_load_to_lds(g0, g1, gz, gz, 0);
#endif
    __builtin_amdgcn_s_wait_tensorcnt(0);
  }
  __syncthreads();
#else
  {
    const int chunksPerCol = K / 8;            // v8h chunks per column
    for (int idx = threadIdx.x; idx < 64 * chunksPerCol; idx += blockDim.x) {
      int c = idx / chunksPerCol, ch = idx % chunksPerCol;
      *(v8h*)&sB[c * KP + ch * 8] = *(const v8h*)&Bt[(size_t)(col0 + c) * K + ch * 8];
    }
    __syncthreads();
  }
#endif

  if (row0 < M) {                              // wave-uniform: EXEC all-ones for WMMA
    v8f acc[4] = {{}, {}, {}, {}};

    for (int k0 = 0; k0 < K; k0 += 32) {
      // A fragment 16x32 f16: lane(hf,lr) holds K = hf*8 + [0..7] and 16 + hf*8 + [0..7]
      const _Float16* pa = A + (size_t)(row0 + lr) * K + k0 + hf * 8;
      v8h alo = *(const v8h*)pa;
      v8h ahi = *(const v8h*)(pa + 16);
      v16h a;
#pragma unroll
      for (int e = 0; e < 8; e++) { a[e] = alo[e]; a[e + 8] = ahi[e]; }
      if (k0 + 32 < K) __builtin_prefetch(pa + 32, 0, 1);    // global_prefetch_b8

#pragma unroll
      for (int t = 0; t < 4; t++) {
        // B fragment 32x16 f16 from LDS: lane holds col = lr, K = hf*8 + [0..7] and 16 + hf*8 + [0..7]
        const _Float16* pb = sB + (size_t)(t * 16 + lr) * KP + k0 + hf * 8;
        v8h blo = *(const v8h*)pb;
        v8h bhi = *(const v8h*)(pb + 16);
        v16h b;
#pragma unroll
        for (int e = 0; e < 8; e++) { b[e] = blo[e]; b[e + 8] = bhi[e]; }
        acc[t] = __builtin_amdgcn_wmma_f32_16x16x32_f16(
            /*neg_a=*/false, a, /*neg_b=*/false, b,
            /*c_mod=*/(short)0, acc[t], /*reuse_a=*/false, /*reuse_b=*/false);
      }
    }

    // C/D layout: element r of v8f -> row = row0 + hf*8 + r, col = col0 + t*16 + lr
#pragma unroll
    for (int t = 0; t < 4; t++) {
      int col = col0 + t * 16 + lr;
#pragma unroll
      for (int r = 0; r < 8; r++) {
        C[(size_t)(row0 + hf * 8 + r) * Ncols + col] = acc[t][r];
      }
    }
  }
}

// ---------- per-node attention logits: a_s/a_d [N*H] ----------
__global__ void node_att_kernel(const float* __restrict__ h,
                                const float* __restrict__ att_src,
                                const float* __restrict__ att_dst,
                                float* __restrict__ a_s, float* __restrict__ a_d,
                                int total /*N*H*/, int H, int C) {
  int t = blockIdx.x * blockDim.x + threadIdx.x;
  if (t >= total) return;
  int n = t / H, hd = t % H;
  const float* hp = h + (size_t)n * H * C + (size_t)hd * C;
  const float* as = att_src + hd * C;
  const float* ad = att_dst + hd * C;
  float s = 0.f, d = 0.f;
  for (int c = 0; c < C; c++) { float v = hp[c]; s += v * as[c]; d += v * ad[c]; }
  a_s[t] = s;
  a_d[t] = d;
}

// ---------- edge pass 1: segment max (monotonic unsigned keys) ----------
__global__ void edge_max_kernel(const int* __restrict__ ei,
                                const float* __restrict__ a_s, const float* __restrict__ a_d,
                                unsigned* __restrict__ mkey, int H) {
  int t = blockIdx.x * blockDim.x + threadIdx.x;
  if (t >= ETOT * H) return;
  int e = t / H, hd = t % H;
  int src = (e < EE) ? ei[e] : (e - EE);
  int dst = (e < EE) ? ei[EE + e] : (e - EE);
  float v = leaky(a_s[src * H + hd] + a_d[dst * H + hd]);
  atomicMax(&mkey[dst * H + hd], f2key(v));
}

// ---------- edge pass 2: exp(e - max) and segment sum ----------
__global__ void edge_expsum_kernel(const int* __restrict__ ei,
                                   const float* __restrict__ a_s, const float* __restrict__ a_d,
                                   const unsigned* __restrict__ mkey,
                                   float* __restrict__ eedge, float* __restrict__ denom, int H) {
  int t = blockIdx.x * blockDim.x + threadIdx.x;
  if (t >= ETOT * H) return;
  int e = t / H, hd = t % H;
  int src = (e < EE) ? ei[e] : (e - EE);
  int dst = (e < EE) ? ei[EE + e] : (e - EE);
  float v = leaky(a_s[src * H + hd] + a_d[dst * H + hd]);
  float m = key2f(mkey[dst * H + hd]);
  float ex = __expf(v - m);
  eedge[t] = ex;
  atomicAdd(&denom[dst * H + hd], ex);
}

// ---------- edge pass 3: alpha-weighted scatter, one wave per edge ----------
__global__ void edge_scatter_kernel(const int* __restrict__ ei,
                                    const float* __restrict__ h,
                                    const float* __restrict__ eedge,
                                    const float* __restrict__ denom,
                                    float* __restrict__ out, int H, int C) {
  int gwave = (blockIdx.x * blockDim.x + threadIdx.x) >> 5;
  int lane = threadIdx.x & 31;
  if (gwave >= ETOT) return;
  int e = gwave;
  int src = (e < EE) ? ei[e] : (e - EE);
  int dst = (e < EE) ? ei[EE + e] : (e - EE);
  int total = H * C;
  for (int base = 0; base < total; base += 32) {
    int idx = base + lane;
    int hd = idx / C;  // uniform per iteration (C is a multiple of 32)
    float alpha = eedge[(size_t)e * H + hd] / (denom[dst * H + hd] + 1e-16f);
    float val = h[(size_t)src * total + idx] * alpha;
    atomicAdd(&out[(size_t)dst * total + idx], val);
  }
}

// ---------- bias + ELU, f16 copy for next GEMM ----------
__global__ void bias_elu_to_f16_kernel(const float* __restrict__ in, const float* __restrict__ bias,
                                       _Float16* __restrict__ out16, int n, int cols) {
  int t = blockIdx.x * blockDim.x + threadIdx.x;
  if (t >= n) return;
  out16[t] = (_Float16)elu(in[t] + bias[t % cols]);
}

__global__ void bias_elu_f32_kernel(float* __restrict__ io, const float* __restrict__ bias,
                                    int n, int cols) {
  int t = blockIdx.x * blockDim.x + threadIdx.x;
  if (t >= n) return;
  io[t] = elu(io[t] + bias[t % cols]);
}

// ---------- global mean pool (atomics) ----------
__global__ void pool_kernel(const float* __restrict__ h, const int* __restrict__ batch,
                            float* __restrict__ sums, float* __restrict__ cnt, int n /*NN*HIDC*/) {
  int t = blockIdx.x * blockDim.x + threadIdx.x;
  if (t >= n) return;
  int node = t / HIDC, c = t % HIDC;
  int g = batch[node];
  atomicAdd(&sums[g * HIDC + c], h[t]);
  if (c == 0) atomicAdd(&cnt[g], 1.0f);
}

// ---------- final FC: out[NG x OUTD] ----------
__global__ void fc_kernel(const float* __restrict__ sums, const float* __restrict__ cnt,
                          const float* __restrict__ fcw, const float* __restrict__ fcb,
                          float* __restrict__ out) {
  int t = blockIdx.x * blockDim.x + threadIdx.x;
  if (t >= NG * OUTD) return;
  int g = t / OUTD, o = t % OUTD;
  float cn = cnt[g];
  if (cn < 1.f) cn = 1.f;
  float acc = 0.f;
  for (int c = 0; c < HIDC; c++) acc += (sums[g * HIDC + c] / cn) * fcw[c * OUTD + o];
  out[t] = acc + fcb[o];
}

// ============================================================================
extern "C" void kernel_launch(void* const* d_in, const int* in_sizes, int n_in,
                              void* d_out, int out_size, void* d_ws, size_t ws_size,
                              hipStream_t stream) {
  (void)in_sizes; (void)n_in; (void)out_size; (void)ws_size;
  const float* x        = (const float*)d_in[0];
  const float* W1       = (const float*)d_in[1];
  const float* att_src1 = (const float*)d_in[2];
  const float* att_dst1 = (const float*)d_in[3];
  const float* b1       = (const float*)d_in[4];
  const float* W2       = (const float*)d_in[5];
  const float* att_src2 = (const float*)d_in[6];
  const float* att_dst2 = (const float*)d_in[7];
  const float* b2       = (const float*)d_in[8];
  const float* fc_w     = (const float*)d_in[9];
  const float* fc_b     = (const float*)d_in[10];
  const int*   ei       = (const int*)d_in[11];
  const int*   batch    = (const int*)d_in[12];
  float* out = (float*)d_out;

  // ---- carve workspace ----
  char* w = (char*)d_ws;
  auto alloc = [&](size_t bytes) -> void* {
    void* p = (void*)w;
    w += (bytes + 255) & ~(size_t)255;
    return p;
  };
  _Float16* xf16   = (_Float16*)alloc((size_t)NN * IND * 2);
  _Float16* W1t    = (_Float16*)alloc((size_t)D1 * IND * 2);
  _Float16* W2t    = (_Float16*)alloc((size_t)HIDC * D1 * 2);
  float*    h1     = (float*)alloc((size_t)NN * D1 * 4);
  float*    out1   = (float*)alloc((size_t)NN * D1 * 4);
  _Float16* x2f16  = (_Float16*)alloc((size_t)NN * D1 * 2);
  float*    h2     = (float*)alloc((size_t)NN * HIDC * 4);
  float*    out2   = (float*)alloc((size_t)NN * HIDC * 4);
  float*    a_s1   = (float*)alloc((size_t)NN * HEADS * 4);
  float*    a_d1   = (float*)alloc((size_t)NN * HEADS * 4);
  unsigned* mkey1  = (unsigned*)alloc((size_t)NN * HEADS * 4);
  float*    denom1 = (float*)alloc((size_t)NN * HEADS * 4);
  float*    eedge1 = (float*)alloc((size_t)ETOT * HEADS * 4);
  float*    a_s2   = (float*)alloc((size_t)NN * 4);
  float*    a_d2   = (float*)alloc((size_t)NN * 4);
  unsigned* mkey2  = (unsigned*)alloc((size_t)NN * 4);
  float*    denom2 = (float*)alloc((size_t)NN * 4);
  float*    eedge2 = (float*)alloc((size_t)ETOT * 4);
  float*    sums   = (float*)alloc((size_t)NG * HIDC * 4);
  float*    cnt    = (float*)alloc((size_t)NG * 4);

  const int B = 256;
  auto blks = [](long long n, int b) { return (unsigned)((n + b - 1) / b); };

  // ---- precision conversion / weight transposes ----
  cast_f16_kernel<<<blks((long long)NN * IND, B), B, 0, stream>>>(x, xf16, NN * IND);
  transpose_f16_kernel<<<blks((long long)IND * D1, B), B, 0, stream>>>(W1, W1t, IND, D1);
  transpose_f16_kernel<<<blks((long long)D1 * HIDC, B), B, 0, stream>>>(W2, W2t, D1, HIDC);

  // ---- GEMM1: h1 = x @ W1  (M=50000, K=128, N=512) ----
  {
    int rowTiles = (NN + 15) / 16;                 // 3125
    dim3 grid(blks(rowTiles, 8), D1 / 64);         // 8 waves/block, 64-col panel per block
    size_t shm = (size_t)64 * (IND + 8) * 2;       // padded B panel
    wmma_gemm_kernel<<<grid, 256, shm, stream>>>(xf16, W1t, h1, NN, IND, D1);
  }

  // ---- layer 1 attention ----
  node_att_kernel<<<blks((long long)NN * HEADS, B), B, 0, stream>>>(
      h1, att_src1, att_dst1, a_s1, a_d1, NN * HEADS, HEADS, HIDC);
  fill_u32_kernel<<<blks((long long)NN * HEADS, B), B, 0, stream>>>(mkey1, NEGINF_KEY, NN * HEADS);
  fill_u32_kernel<<<blks((long long)NN * HEADS, B), B, 0, stream>>>((unsigned*)denom1, 0u, NN * HEADS);
  fill_u32_kernel<<<blks((long long)NN * D1, B), B, 0, stream>>>((unsigned*)out1, 0u, NN * D1);
  edge_max_kernel<<<blks((long long)ETOT * HEADS, B), B, 0, stream>>>(ei, a_s1, a_d1, mkey1, HEADS);
  edge_expsum_kernel<<<blks((long long)ETOT * HEADS, B), B, 0, stream>>>(
      ei, a_s1, a_d1, mkey1, eedge1, denom1, HEADS);
  edge_scatter_kernel<<<blks((long long)ETOT * 32, B), B, 0, stream>>>(
      ei, h1, eedge1, denom1, out1, HEADS, HIDC);
  bias_elu_to_f16_kernel<<<blks((long long)NN * D1, B), B, 0, stream>>>(out1, b1, x2f16, NN * D1, D1);

  // ---- GEMM2: h2 = elu(out1+b1) @ W2  (M=50000, K=512, N=64) ----
  {
    int rowTiles = (NN + 15) / 16;
    dim3 grid(blks(rowTiles, 8), HIDC / 64);
    size_t shm = (size_t)64 * (D1 + 8) * 2;
    wmma_gemm_kernel<<<grid, 256, shm, stream>>>(x2f16, W2t, h2, NN, D1, HIDC);
  }

  // ---- layer 2 attention (H=1) ----
  node_att_kernel<<<blks(NN, B), B, 0, stream>>>(h2, att_src2, att_dst2, a_s2, a_d2, NN, 1, HIDC);
  fill_u32_kernel<<<blks(NN, B), B, 0, stream>>>(mkey2, NEGINF_KEY, NN);
  fill_u32_kernel<<<blks(NN, B), B, 0, stream>>>((unsigned*)denom2, 0u, NN);
  fill_u32_kernel<<<blks((long long)NN * HIDC, B), B, 0, stream>>>((unsigned*)out2, 0u, NN * HIDC);
  edge_max_kernel<<<blks(ETOT, B), B, 0, stream>>>(ei, a_s2, a_d2, mkey2, 1);
  edge_expsum_kernel<<<blks(ETOT, B), B, 0, stream>>>(ei, a_s2, a_d2, mkey2, eedge2, denom2, 1);
  edge_scatter_kernel<<<blks((long long)ETOT * 32, B), B, 0, stream>>>(
      ei, h2, eedge2, denom2, out2, 1, HIDC);
  bias_elu_f32_kernel<<<blks((long long)NN * HIDC, B), B, 0, stream>>>(out2, b2, NN * HIDC, HIDC);

  // ---- global mean pool + FC ----
  fill_u32_kernel<<<blks(NG * HIDC, B), B, 0, stream>>>((unsigned*)sums, 0u, NG * HIDC);
  fill_u32_kernel<<<1, NG, 0, stream>>>((unsigned*)cnt, 0u, NG);
  pool_kernel<<<blks((long long)NN * HIDC, B), B, 0, stream>>>(out2, batch, sums, cnt, NN * HIDC);
  fc_kernel<<<1, 128, 0, stream>>>(sums, cnt, fc_w, fc_b, out);
}
